// SRPWCNet_26826365731221
// MI455X (gfx1250) — compile-verified, matrix operand order checked by reading, not run on hardware
//
#include <hip/hip_runtime.h>
#include <cstdint>
#include <cstddef>

// ---------------------------------------------------------------------------
// Types
// ---------------------------------------------------------------------------
typedef __bf16 v16bf __attribute__((ext_vector_type(16)));
typedef __bf16 v8bf  __attribute__((ext_vector_type(8)));
typedef float  v8f   __attribute__((ext_vector_type(8)));
typedef int    v4i   __attribute__((ext_vector_type(4)));

#if defined(__has_builtin)
#if __has_builtin(__builtin_amdgcn_global_load_async_to_lds_b128) && \
    __has_builtin(__builtin_amdgcn_s_wait_asynccnt)
#define HAS_ASYNC_LDS 1
#endif
#endif

__device__ __forceinline__ float bf2f(unsigned short h) {
  unsigned int u = ((unsigned int)h) << 16;
  return __builtin_bit_cast(float, u);
}
__device__ __forceinline__ unsigned short f2bf(float f) {
  unsigned int u = __builtin_bit_cast(unsigned int, f);
  u += 0x7fffu + ((u >> 16) & 1u);   // round-to-nearest-even
  return (unsigned short)(u >> 16);
}

#if HAS_ASYNC_LDS
// arg0 = v4i32 addrspace(1)* (global), arg1 = v4i32 addrspace(3)* (LDS),
// then imm offset, imm cpol (verified against hipcc diagnostics).
__device__ __forceinline__ void g2l_async_b128(const void* g, void* l) {
  __builtin_amdgcn_global_load_async_to_lds_b128(
      (__attribute__((address_space(1))) v4i*)g,
      (__attribute__((address_space(3))) v4i*)l, 0, 0);
}
#endif

// Load a 16x32 bf16 WMMA fragment row (A layout, ISA 7.12.2): lane L<16 holds
// row M=L with K 0..7 / 16..23; lanes 16..31 hold K 8..15 / 24..31.
__device__ __forceinline__ v16bf ldfrag(const unsigned short* row, int half) {
  v8bf lo = *(const v8bf*)(row + half * 8);
  v8bf hi = *(const v8bf*)(row + 16 + half * 8);
  return __builtin_shufflevector(lo, hi, 0,1,2,3,4,5,6,7,8,9,10,11,12,13,14,15);
}

// ---------------------------------------------------------------------------
// Implicit-GEMM 3x3 conv (stride/dilation), NHWC bf16 in, bf16/f32 out.
//   M = Hout*Wout pixels (tile 128), N = Cout (tile 64), K = 9*CinPad (step 32)
//   256 threads = 8 waves; wave(wm,wn) computes 32x32 via 2x2 WMMA 16x16x32.
//   Double-buffered LDS staging; async global->LDS with deterministic count
//   (3 per stage) so s_wait_asynccnt(3) pipelines next-tile loads behind WMMA.
// ---------------------------------------------------------------------------
#define MT 128
#define NT 64
#define AP 40   // LDS row stride in bf16 elements (32 data + 8 pad)

__global__ __launch_bounds__(256, 2)
void conv3x3_wmma(const unsigned short* __restrict__ in,
                  const unsigned short* __restrict__ wgt,   // [CoutPad][9*CinPad]
                  const float* __restrict__ bias,
                  unsigned short* __restrict__ out,         // NHWC, OutCstride
                  float* __restrict__ fout,                 // NCHW f32 (Cout==2)
                  int Hin, int Win, int CinPad,
                  int Hout, int Wout, int Cout,
                  int OutCstride, int OutCoff,
                  int stride, int dil, int leaky)
{
  __shared__ unsigned short ldsA[2][MT * AP];
  __shared__ unsigned short ldsB[2][NT * AP];
  __shared__ unsigned short trash[32];      // async dump for padded lanes

  const int t     = threadIdx.x;
  const int lane  = t & 31, wave = t >> 5;
  const int waveM = wave & 3, waveN = wave >> 2;
  const int ln    = lane & 15, half = lane >> 4;
  const int tileM = blockIdx.x, tileN = blockIdx.y, bz = blockIdx.z;
  const int Mtot  = Hout * Wout;

  // A-staging: thread -> (pixel pA, 16-elem chunk ca); 2x b128 per thread.
  const int pA  = t >> 1;
  const int ca  = (t & 1) * 16;
  const int mA  = tileM * MT + pA;
  const bool mval = mA < Mtot;
  const int oyS = (mval ? mA / Wout : 0) * stride;
  const int oxS = (mval ? mA % Wout : 0) * stride;
  const unsigned short* inb = in + (size_t)bz * Hin * Win * CinPad;

  // B-staging: thread -> (row nB, 8-elem chunk cb); 1x b128 per thread.
  const int nB  = t >> 2;
  const int cb  = (t & 3) * 8;
  const int Ktot = 9 * CinPad;

  v8f acc[2][2] = {};

  const int cinT   = CinPad >> 5;
  const int ktiles = 9 * cinT;

  auto stage = [&](int kt, int buf) {
    const int tap = kt / cinT;
    const int ck  = (kt - tap * cinT) << 5;
    // ---- A tile (activation gather, spatial zero-padding)
    const int iy = oyS + dil * (tap / 3 - 1);
    const int ix = oxS + dil * (tap % 3 - 1);
    const bool v = mval && (unsigned)iy < (unsigned)Hin &&
                           (unsigned)ix < (unsigned)Win;
    const int iyc = min(max(iy, 0), Hin - 1);
    const int ixc = min(max(ix, 0), Win - 1);
    uint4* lpA = (uint4*)&ldsA[buf][pA * AP + ca];
    const uint4* gpA = (const uint4*)(inb + ((size_t)iyc * Win + ixc) * CinPad + ck + ca);
#if HAS_ASYNC_LDS
    if (!v) {                      // zero real slot; async goes to trash
      uint4 z = {0u, 0u, 0u, 0u};
      lpA[0] = z; lpA[1] = z;
    }
    uint4* dstA = v ? lpA : (uint4*)&trash[0];
    g2l_async_b128(gpA,     dstA);      // always 2 async ops: deterministic
    g2l_async_b128(gpA + 1, dstA + 1);
#else
    if (v) { lpA[0] = gpA[0]; lpA[1] = gpA[1]; }
    else   { uint4 z = {0u, 0u, 0u, 0u}; lpA[0] = z; lpA[1] = z; }
#endif
    // ---- B tile (pre-transposed weights [n][k], always in-bounds)
    const uint4* gpB = (const uint4*)
        (wgt + (size_t)(tileN * NT + nB) * Ktot + tap * CinPad + ck + cb);
    uint4* lpB = (uint4*)&ldsB[buf][nB * AP + cb];
#if HAS_ASYNC_LDS
    g2l_async_b128(gpB, lpB);           // 3rd async op
#else
    lpB[0] = gpB[0];
#endif
  };

  stage(0, 0);
  for (int kt = 0; kt < ktiles; ++kt) {
    const int cur = kt & 1;
    const bool havenext = (kt + 1) < ktiles;
    if (havenext) stage(kt + 1, cur ^ 1);   // prefetch next tile (3 async ops)
#if HAS_ASYNC_LDS
    if (havenext) __builtin_amdgcn_s_wait_asynccnt(3);  // drain current tile only
    else          __builtin_amdgcn_s_wait_asynccnt(0);
#endif
    __syncthreads();

    v16bf a0 = ldfrag(&ldsA[cur][(waveM * 32 +      ln) * AP], half);
    v16bf a1 = ldfrag(&ldsA[cur][(waveM * 32 + 16 + ln) * AP], half);
    v16bf b0 = ldfrag(&ldsB[cur][(waveN * 32 +      ln) * AP], half);
    v16bf b1 = ldfrag(&ldsB[cur][(waveN * 32 + 16 + ln) * AP], half);

    acc[0][0] = __builtin_amdgcn_wmma_f32_16x16x32_bf16(
        false, a0, false, b0, (short)0, acc[0][0], false, false);
    acc[0][1] = __builtin_amdgcn_wmma_f32_16x16x32_bf16(
        false, a0, false, b1, (short)0, acc[0][1], false, false);
    acc[1][0] = __builtin_amdgcn_wmma_f32_16x16x32_bf16(
        false, a1, false, b0, (short)0, acc[1][0], false, false);
    acc[1][1] = __builtin_amdgcn_wmma_f32_16x16x32_bf16(
        false, a1, false, b1, (short)0, acc[1][1], false, false);

    __syncthreads();   // all waves done reading buf[cur] before it is restaged
  }

  // Epilogue. D layout: VGPR r -> lanes 0-15 M=r, lanes 16-31 M=r+8; N=lane&15.
#pragma unroll
  for (int sm = 0; sm < 2; ++sm) {
#pragma unroll
    for (int sn = 0; sn < 2; ++sn) {
      const int cc = tileN * NT + waveN * 32 + sn * 16 + ln;
      const float bv = (cc < Cout) ? bias[cc] : 0.f;
      v8f a = acc[sm][sn];
#pragma unroll
      for (int r = 0; r < 8; ++r) {
        const int mm = tileM * MT + waveM * 32 + sm * 16 + r + half * 8;
        if (mm >= Mtot) continue;
        float v = a[r] + bv;
        if (leaky) v = (v > 0.f) ? v : 0.1f * v;
        const int oy = mm / Wout, ox = mm % Wout;
        if (out && cc < Cout)
          out[(((size_t)bz * Hout + oy) * Wout + ox) * OutCstride + OutCoff + cc] = f2bf(v);
        if (fout && cc < 2)
          fout[(((size_t)bz * 2 + cc) * Hout + oy) * Wout + ox] = v;
      }
    }
  }
}

// ---------------------------------------------------------------------------
// Weight transform: OIHW f32 -> [CoutPad][tap][CinPad] bf16, zero padded.
// ---------------------------------------------------------------------------
__global__ void prep_w(const float* __restrict__ w, unsigned short* __restrict__ o,
                       int Cout, int Cin, int CinPad, int CoutPad)
{
  int idx = blockIdx.x * 256 + threadIdx.x;
  int total = CoutPad * 9 * CinPad;
  if (idx >= total) return;
  int ci  = idx % CinPad;
  int tap = (idx / CinPad) % 9;
  int co  = idx / (CinPad * 9);
  float v = 0.f;
  if (co < Cout && ci < Cin) {
    int ky = tap / 3, kx = tap % 3;
    v = w[((co * Cin + ci) * 3 + ky) * 3 + kx];
  }
  o[idx] = f2bf(v);
}

// ---------------------------------------------------------------------------
// First conv (Cin=3, stride 2) direct: NCHW f32 image -> NHWC bf16 (Cpad=32).
// ---------------------------------------------------------------------------
__global__ void conv0_direct(const float* __restrict__ im, const float* __restrict__ w,
                             const float* __restrict__ bias, unsigned short* __restrict__ out,
                             int Hin, int Win, int Hout, int Wout)
{
  int idx = blockIdx.x * 256 + threadIdx.x;
  int total = 4 * Hout * Wout * 32;
  if (idx >= total) return;
  int co = idx & 31;
  int p  = idx >> 5;
  int ox = p % Wout, oy = (p / Wout) % Hout, b = p / (Wout * Hout);
  float s = bias[co];
  for (int ci = 0; ci < 3; ++ci)
    for (int ky = 0; ky < 3; ++ky) {
      int iy = oy * 2 + ky - 1;
      if ((unsigned)iy >= (unsigned)Hin) continue;
      for (int kx = 0; kx < 3; ++kx) {
        int ix = ox * 2 + kx - 1;
        if ((unsigned)ix >= (unsigned)Win) continue;
        s += im[(((size_t)b * 3 + ci) * Hin + iy) * Win + ix] *
             w[((co * 3 + ci) * 3 + ky) * 3 + kx];
      }
    }
  s = (s > 0.f) ? s : 0.1f * s;
  out[(size_t)p * 32 + co] = f2bf(s);
}

// ---------------------------------------------------------------------------
// Misc small kernels
// ---------------------------------------------------------------------------
__global__ void zero_f32(float* p, int n) {
  int i = blockIdx.x * 256 + threadIdx.x;
  if (i < n) p[i] = 0.f;
}

__global__ void upsample2(const float* __restrict__ in, float* __restrict__ out,
                          int h, int w)  // (B,2,h,w) -> (B,2,2h,2w), bilinear
{
  int H = 2 * h, W = 2 * w;
  int idx = blockIdx.x * 256 + threadIdx.x;
  int total = 4 * 2 * H * W;
  if (idx >= total) return;
  int x = idx % W, y = (idx / W) % H, ch = (idx / (W * H)) & 1, b = idx / (W * H * 2);
  float sy = y * 0.5f - 0.25f, sx = x * 0.5f - 0.25f;
  int y0 = (int)floorf(sy), x0 = (int)floorf(sx);
  float wy = sy - (float)y0, wx = sx - (float)x0;
  int y0c = min(max(y0, 0), h - 1), y1c = min(max(y0 + 1, 0), h - 1);
  int x0c = min(max(x0, 0), w - 1), x1c = min(max(x0 + 1, 0), w - 1);
  const float* p = in + (size_t)(b * 2 + ch) * h * w;
  out[idx] = (1.f - wy) * ((1.f - wx) * p[y0c * w + x0c] + wx * p[y0c * w + x1c]) +
             wy        * ((1.f - wx) * p[y1c * w + x0c] + wx * p[y1c * w + x1c]);
}

__global__ void warp_k(const unsigned short* __restrict__ feat,
                       const float* __restrict__ flow, float scale,
                       unsigned short* __restrict__ out, int H, int W, int CP, int C)
{
  int idx = blockIdx.x * 256 + threadIdx.x;
  int total = 4 * H * W;
  if (idx >= total) return;
  int x = idx % W, y = (idx / W) % H, b = idx / (H * W);
  float fx = (float)x + scale * flow[(((size_t)b * 2 + 0) * H + y) * W + x];
  float fy = (float)y + scale * flow[(((size_t)b * 2 + 1) * H + y) * W + x];
  float fx0 = floorf(fx), fy0 = floorf(fy);
  int x0 = (int)fx0, y0 = (int)fy0;
  float wx = fx - fx0, wy = fy - fy0;
  float wg[2][2] = {{(1.f - wx) * (1.f - wy), wx * (1.f - wy)},
                    {(1.f - wx) * wy,         wx * wy}};
  const unsigned short* rows[2][2];
  float wv[2][2];
#pragma unroll
  for (int iy = 0; iy < 2; ++iy)
#pragma unroll
    for (int ix = 0; ix < 2; ++ix) {
      int yy = y0 + iy, xx = x0 + ix;
      bool v = (yy >= 0 && yy < H && xx >= 0 && xx < W);
      int yc = min(max(yy, 0), H - 1), xc = min(max(xx, 0), W - 1);
      rows[iy][ix] = feat + (((size_t)b * H + yc) * W + xc) * CP;
      wv[iy][ix] = v ? wg[iy][ix] : 0.f;
    }
  unsigned short* op = out + (size_t)idx * CP;
  for (int c = 0; c < CP; ++c) {
    float s = 0.f;
    if (c < C)
      s = wv[0][0] * bf2f(rows[0][0][c]) + wv[0][1] * bf2f(rows[0][1][c]) +
          wv[1][0] * bf2f(rows[1][0][c]) + wv[1][1] * bf2f(rows[1][1][c]);
    op[c] = f2bf(s);
  }
}

__global__ void corr81(const unsigned short* __restrict__ f1,
                       const unsigned short* __restrict__ f2,
                       unsigned short* __restrict__ xf,
                       int H, int W, int C, int Cpad, int XfPad)
{
  int idx = blockIdx.x * 256 + threadIdx.x;
  int total = 4 * H * W;
  if (idx >= total) return;
  int x = idx % W, y = (idx / W) % H, b = idx / (H * W);
  float acc[81];
#pragma unroll
  for (int i = 0; i < 81; ++i) acc[i] = 0.f;
  const unsigned short* p1 = f1 + (size_t)idx * Cpad;
  for (int c = 0; c < C; ++c) {
    float a = bf2f(p1[c]);
#pragma unroll
    for (int dy = 0; dy < 9; ++dy) {
      int yy = y + dy - 4;
      if ((unsigned)yy >= (unsigned)H) continue;
      const unsigned short* r2 = f2 + (((size_t)b * H + yy) * W) * Cpad + c;
#pragma unroll
      for (int dx = 0; dx < 9; ++dx) {
        int xx = x + dx - 4;
        if ((unsigned)xx < (unsigned)W) acc[dy * 9 + dx] += a * bf2f(r2[(size_t)xx * Cpad]);
      }
    }
  }
  float inv = 1.f / (float)C;
  unsigned short* op = xf + (size_t)idx * XfPad;
#pragma unroll
  for (int i = 0; i < 81; ++i) {
    float m = acc[i] * inv;
    m = (m > 0.f) ? m : 0.1f * m;   // leaky relu fused on corr
    op[i] = f2bf(m);
  }
}

__global__ void copy_feat(const unsigned short* __restrict__ src,
                          unsigned short* __restrict__ dst,
                          int npx, int C, int srcCP, int dstCP, int dstOff)
{
  int idx = blockIdx.x * 256 + threadIdx.x;
  if (idx >= npx * C) return;
  int c = idx % C, p = idx / C;
  dst[(size_t)p * dstCP + dstOff + c] = src[(size_t)p * srcCP + c];
}

__global__ void copy_flow(const float* __restrict__ flow, unsigned short* __restrict__ dst,
                          int H, int W, int dstCP, int dstOff)
{
  int idx = blockIdx.x * 256 + threadIdx.x;
  int total = 4 * H * W * 2;
  if (idx >= total) return;
  int ch = idx & 1, p = idx >> 1;
  int x = p % W, y = (p / W) % H, b = p / (H * W);
  dst[(size_t)p * dstCP + dstOff + ch] = f2bf(flow[(((size_t)b * 2 + ch) * H + y) * W + x]);
}

__global__ void final_add(const float* __restrict__ a, const float* __restrict__ b,
                          float* __restrict__ o, int n)
{
  int i = blockIdx.x * 256 + threadIdx.x;
  if (i < n) o[i] = a[i] + b[i];
}

// ---------------------------------------------------------------------------
// Host orchestration
// ---------------------------------------------------------------------------
static void run_conv(hipStream_t s, unsigned short* wbuf,
                     const unsigned short* in, const void* w_raw, const void* bias,
                     unsigned short* out, float* fout,
                     int Hin, int Win, int CinPad, int Cin,
                     int Hout, int Wout, int Cout, int OutCs, int OutCo,
                     int stride, int dil, int leaky)
{
  int CoutPad = ((Cout + 63) / 64) * 64;
  int n = CoutPad * 9 * CinPad;
  prep_w<<<(n + 255) / 256, 256, 0, s>>>((const float*)w_raw, wbuf, Cout, Cin, CinPad, CoutPad);
  dim3 g((Hout * Wout + MT - 1) / MT, (Cout + NT - 1) / NT, 4);
  conv3x3_wmma<<<g, 256, 0, s>>>(in, wbuf, (const float*)bias, out, fout,
                                 Hin, Win, CinPad, Hout, Wout, Cout,
                                 OutCs, OutCo, stride, dil, leaky);
}

extern "C" void kernel_launch(void* const* d_in, const int* in_sizes, int n_in,
                              void* d_out, int out_size, void* d_ws, size_t ws_size,
                              hipStream_t stream)
{
  (void)in_sizes; (void)n_in; (void)ws_size;
  const float* im1 = (const float*)d_in[0];
  const float* im2 = (const float*)d_in[1];

  // params dict leaves assumed flattened in sorted-key (jax pytree) order:
  // cb0..cb5, cbf, cw0..cw5, cwf, [e{l}: b0..b4, bf, w0..w4, wf]*4, fb0..4, fw0..4
  auto PAR  = [&](int i) { return d_in[2 + i]; };
  auto c_b  = [&](int i) { return PAR(i); };
  auto c_bf = [&]()      { return PAR(6); };
  auto c_w  = [&](int i) { return PAR(7 + i); };
  auto c_wf = [&]()      { return PAR(13); };
  auto e_b  = [&](int l, int i) { return PAR(14 + l * 12 + i); };
  auto e_bf = [&](int l)        { return PAR(14 + l * 12 + 5); };
  auto e_w  = [&](int l, int i) { return PAR(14 + l * 12 + 6 + i); };
  auto e_wf = [&](int l)        { return PAR(14 + l * 12 + 11); };
  auto f_b  = [&](int i) { return PAR(62 + i); };
  auto f_w  = [&](int i) { return PAR(67 + i); };

  const int Hs[5] = {224, 112, 56, 28, 14};
  const int Ws[5] = {512, 256, 128, 64, 32};
  const int Cs[5] = {32, 64, 96, 128, 196};
  const int CPs[5] = {32, 64, 96, 128, 224};

  // bump allocator over d_ws
  char* wsb = (char*)d_ws;
  size_t off = 0;
  auto alloc = [&](size_t bytes) -> void* {
    void* p = wsb + off;
    off += (bytes + 255) & ~(size_t)255;
    return p;
  };

  unsigned short *feat1[5], *feat2[5];
  for (int i = 0; i < 5; ++i)
    feat1[i] = (unsigned short*)alloc((size_t)4 * Hs[i] * Ws[i] * CPs[i] * 2);
  for (int i = 0; i < 5; ++i)
    feat2[i] = (unsigned short*)alloc((size_t)4 * Hs[i] * Ws[i] * CPs[i] * 2);
  unsigned short* warpb = (unsigned short*)alloc((size_t)4 * 112 * 256 * 64 * 2);
  unsigned short* xf    = (unsigned short*)alloc((size_t)4 * 112 * 256 * 160 * 2);
  unsigned short* hidA  = (unsigned short*)alloc((size_t)4 * 112 * 256 * 128 * 2);
  unsigned short* hidB  = (unsigned short*)alloc((size_t)4 * 112 * 256 * 128 * 2);
  unsigned short* ctxin = (unsigned short*)alloc((size_t)4 * 112 * 256 * 64 * 2);
  float* flowA = (float*)alloc((size_t)4 * 2 * 112 * 256 * 4);
  float* flowB = (float*)alloc((size_t)4 * 2 * 112 * 256 * 4);
  float* upfl  = (float*)alloc((size_t)4 * 2 * 112 * 256 * 4);
  float* resid = (float*)alloc((size_t)4 * 2 * 112 * 256 * 4);
  unsigned short* wbuf = (unsigned short*)alloc((size_t)1 << 20);

  // ---- feature pyramids (both images) ----
  for (int img = 0; img < 2; ++img) {
    const float* im = img ? im2 : im1;
    unsigned short** F = img ? feat2 : feat1;
    int n0 = 4 * Hs[0] * Ws[0] * 32;
    conv0_direct<<<(n0 + 255) / 256, 256, 0, stream>>>(
        im, (const float*)f_w(0), (const float*)f_b(0), F[0], 448, 1024, Hs[0], Ws[0]);
    for (int i = 1; i < 5; ++i)
      run_conv(stream, wbuf, F[i - 1], f_w(i), f_b(i), F[i], nullptr,
               Hs[i - 1], Ws[i - 1], CPs[i - 1], Cs[i - 1],
               Hs[i], Ws[i], Cs[i], CPs[i], 0, /*stride*/2, /*dil*/1, /*leaky*/1);
  }

  // ---- coarse-to-fine flow estimation ----
  const float scales[4] = {0.625f, 1.25f, 2.5f, 5.0f};
  const int featC[4]  = {196, 128, 96, 64};
  const int featCP[4] = {224, 128, 96, 64};
  const int lH[4] = {14, 28, 56, 112};
  const int lW[4] = {32, 64, 128, 256};
  const int xfPadA[4] = {288, 224, 192, 160};
  const int xfCinA[4] = {277, 211, 179, 147};

  float* flow_prev = nullptr;
  for (int l = 0; l < 4; ++l) {
    const int H = lH[l], W = lW[l], C = featC[l], CP = featCP[l];
    const int np = 4 * H * W;
    const int pl = 4 - l;  // pyramid index

    if (l == 0)
      zero_f32<<<(4 * 2 * H * W + 255) / 256, 256, 0, stream>>>(upfl, 4 * 2 * H * W);
    else
      upsample2<<<(4 * 2 * H * W + 255) / 256, 256, 0, stream>>>(flow_prev, upfl, H / 2, W / 2);

    warp_k<<<(np + 255) / 256, 256, 0, stream>>>(feat2[pl], upfl, scales[l], warpb, H, W, CP, C);
    corr81<<<(np + 255) / 256, 256, 0, stream>>>(feat1[pl], warpb, xf, H, W, C, CP, xfPadA[l]);
    copy_feat<<<(np * C + 255) / 256, 256, 0, stream>>>(feat1[pl], xf, np, C, CP, xfPadA[l], 81);
    if (l > 0)
      copy_flow<<<(np * 2 + 255) / 256, 256, 0, stream>>>(upfl, xf, H, W, xfPadA[l], 81 + C);

    const int ecin[5]  = {xfCinA[l], 128, 128, 96, 64};
    const int ecinP[5] = {xfPadA[l], 128, 128, 96, 64};
    const int ecout[5] = {128, 128, 96, 64, 32};
    const unsigned short* src = xf;
    unsigned short* dsts[5] = {hidA, hidB, hidA, hidB, hidA};
    for (int i = 0; i < 5; ++i) {
      run_conv(stream, wbuf, src, e_w(l, i), e_b(l, i), dsts[i], nullptr,
               H, W, ecinP[i], ecin[i], H, W, ecout[i], ecout[i], 0, 1, 1, 1);
      src = dsts[i];
    }
    float* flow_cur = (l & 1) ? flowB : flowA;
    run_conv(stream, wbuf, hidA, e_wf(l), e_bf(l), nullptr, flow_cur,
             H, W, 32, 32, H, W, 2, 1, 0, 1, 1, /*leaky*/0);
    flow_prev = flow_cur;  // flow_features for l==3 stays in hidA
  }

  // ---- context network at finest level (112x256) ----
  {
    const int H = 112, W = 256, np = 4 * H * W;
    copy_flow<<<(np * 2 + 255) / 256, 256, 0, stream>>>(flow_prev, ctxin, H, W, 64, 0);
    copy_feat<<<(np * 32 + 255) / 256, 256, 0, stream>>>(hidA, ctxin, np, 32, 32, 64, 2);

    const int ccin[6]  = {34, 128, 128, 128, 96, 64};
    const int ccinP[6] = {64, 128, 128, 128, 96, 64};
    const int ccout[6] = {128, 128, 128, 96, 64, 32};
    const int cdil[6]  = {1, 2, 4, 8, 16, 1};
    const unsigned short* src = ctxin;
    unsigned short* dsts[6] = {hidA, hidB, hidA, hidB, hidA, hidB};
    for (int i = 0; i < 6; ++i) {
      run_conv(stream, wbuf, src, c_w(i), c_b(i), dsts[i], nullptr,
               H, W, ccinP[i], ccin[i], H, W, ccout[i], ccout[i], 0, 1, cdil[i], 1);
      src = dsts[i];
    }
    run_conv(stream, wbuf, hidB, c_wf(), c_bf(), nullptr, resid,
             H, W, 32, 32, H, W, 2, 1, 0, 1, 1, /*leaky*/0);

    final_add<<<(out_size + 255) / 256, 256, 0, stream>>>(flow_prev, resid, (float*)d_out, out_size);
  }
}
